// S2ConvNet_38182259261583
// MI455X (gfx1250) — matrix operand order repaired and checked by reading, not compiled, not run on hardware
//
#include <hip/hip_runtime.h>

typedef __attribute__((ext_vector_type(16))) _Float16 v16h;
typedef __attribute__((ext_vector_type(8)))  float    v8f;

#define PI_D 3.14159265358979323846
#define TWOPI_F 6.28318530717958647692f

// ---------------- workspace layout ----------------
// double region: (double*)d_ws, 272 doubles used, padded to 640 floats
//   LF[64] @0, b0[64] @64, w0[64] @128, b1[32] @192, w1q[32] @224, b2[16] @256
// float region F = (float*)d_ws + 640:
constexpr size_t OFF_D1F  = 0;          // 16*64*31        = 31744
constexpr size_t OFF_D1I  = 31744;      // 16*32*31*31     = 492032
constexpr size_t OFF_D2F  = 523776;     // 8*32*15*15      = 57600
constexpr size_t OFF_D2I0 = 581376;     // 8*15*15         = 1800
constexpr size_t OFF_XMR  = 583176;     // 16*64*31
constexpr size_t OFF_XMI  = 614920;
constexpr size_t OFF_XHR  = 646664;     // 16*16*31
constexpr size_t OFF_XHI  = 654600;
constexpr size_t OFF_FEAT = 662536;     // 16*64
constexpr size_t OFF_SR   = 663584;     // 16*32*32*32*15  = 7864320
constexpr size_t OFF_SI   = 8527904;
constexpr size_t OFF_XMNR = 16392224;   // 16*32*32*15*15  = 3686400
constexpr size_t OFF_XMNI = 20078624;
constexpr size_t OFF_X2R  = 23765024;   // 16*32*8*15*15   = 921600
constexpr size_t OFF_X2I  = 24686624;
constexpr size_t OFF_XP   = 25608224;   // _Float16 region: 1843200 halves
constexpr size_t OFF_WP   = 26529824;   // _Float16 region: 7372800 halves
// total floats = 30216224  (~115 MiB + 640-float header)

// ---------------- device helpers ----------------
__device__ __forceinline__ double dev_ipow(double b, int e) {
    double r = 1.0;
    for (int i = 0; i < e; ++i) r *= b;
    return r;
}

__device__ double dev_wig(int l, int m, int n, double beta, const double* LF) {
    int kmin = (m - n > 0) ? (m - n) : 0;
    int kmax = (l + m < l - n) ? (l + m) : (l - n);
    double pref = 0.5 * (LF[l + m] + LF[l - m] + LF[l + n] + LF[l - n]);
    double cb = cos(0.5 * beta), sb = sin(0.5 * beta);
    double acc = 0.0;
    for (int k = kmin; k <= kmax; ++k) {
        double c = exp(pref - LF[k] - LF[l + m - k] - LF[l - n - k] - LF[n - m + k]);
        if (k & 1) c = -c;
        acc += c * dev_ipow(cb, 2 * l + m - n - 2 * k) * dev_ipow(sb, n - m + 2 * k);
    }
    return acc;
}

// ---------------- constant builders ----------------
__global__ void k_scalars(double* Dd) {
    int t = threadIdx.x;
    double* LF = Dd;          double* b0 = Dd + 64;  double* w0 = Dd + 128;
    double* b1 = Dd + 192;    double* w1 = Dd + 224; double* b2 = Dd + 256;
    if (t < 64) {
        double a = 0.0;
        for (int i = 2; i <= t; ++i) a += log((double)i);
        LF[t] = a;
        double beta = PI_D * (2.0 * t + 1.0) / 128.0;  // B0 = 32
        b0[t] = beta;
        double s = 0.0;
        for (int k = 0; k < 32; ++k) s += sin((2.0 * k + 1.0) * beta) / (2.0 * k + 1.0);
        w0[t] = (2.0 / 32.0) * sin(beta) * s;
    }
    if (t < 32) {
        double beta = PI_D * (2.0 * t + 1.0) / 64.0;   // B1 = 16
        b1[t] = beta;
        double s = 0.0;
        for (int k = 0; k < 16; ++k) s += sin((2.0 * k + 1.0) * beta) / (2.0 * k + 1.0);
        w1[t] = (2.0 / 16.0) * sin(beta) * s;
    }
    if (t < 16) b2[t] = PI_D * (2.0 * t + 1.0) / 32.0; // B2 = 8
}

__global__ void k_zero(float* p, int n) {
    int i = blockIdx.x * 256 + threadIdx.x;
    if (i < n) p[i] = 0.0f;
}

__global__ void k_d1f(const double* Dd, float* F) {
    int i = blockIdx.x * 256 + threadIdx.x;            // < 31744
    int mi = i % 31, j = (i / 31) & 63, l = i / 1984;
    int m = mi - 15;
    if (m < -l || m > l) return;
    const double* LF = Dd; const double* b0 = Dd + 64; const double* w0 = Dd + 128;
    F[OFF_D1F + (size_t)(l * 64 + j) * 31 + mi] = (float)(w0[j] * dev_wig(l, m, 0, b0[j], LF));
}

__global__ void k_d1i(const double* Dd, float* F) {
    int i = blockIdx.x * 256 + threadIdx.x;            // < 492032
    int ni = i % 31, mi = (i / 31) % 31, j = (i / 961) & 31, l = i / 30752;
    int m = mi - 15, n = ni - 15;
    if (m < -l || m > l || n < -l || n > l) return;
    const double* LF = Dd; const double* b1 = Dd + 192;
    F[OFF_D1I + ((size_t)(l * 32 + j) * 31 + mi) * 31 + ni] =
        (float)((2.0 * l + 1.0) * dev_wig(l, m, n, b1[j], LF));
}

__global__ void k_d2f(const double* Dd, float* F) {
    int i = blockIdx.x * 256 + threadIdx.x;            // < 57600
    int ki = i % 15, mi = (i / 15) % 15, j = (i / 225) & 31, l = i / 7200;
    int m = mi - 7, n = ki - 7;
    if (m < -l || m > l || n < -l || n > l) return;
    const double* LF = Dd; const double* b1 = Dd + 192; const double* w1 = Dd + 224;
    F[OFF_D2F + ((size_t)(l * 32 + j) * 15 + mi) * 15 + ki] =
        (float)(w1[j] * dev_wig(l, m, n, b1[j], LF));
}

__global__ void k_d2i0(const double* Dd, float* F) {
    int i = blockIdx.x * 256 + threadIdx.x;            // < 1800
    if (i >= 1800) return;
    int ni = i % 15, mi = (i / 15) % 15, l = i / 225;
    int m = mi - 7, n = ni - 7;
    if (m < -l || m > l || n < -l || n > l) return;
    const double* LF = Dd; const double* b2 = Dd + 256;
    F[OFF_D2I0 + (size_t)(l * 15 + mi) * 15 + ni] =
        (float)((2.0 * l + 1.0) * dev_wig(l, m, n, b2[0], LF));
}

// ---------------- forward pipeline ----------------
// xm[b,j,mi] = sum_t x[b,0,j,t] * e^{-2pi i (mi-15) t / 64}
__global__ void k_fft1(const float* __restrict__ x, float* F) {
    __shared__ float c64[64], s64[64];
    int tid = threadIdx.x;
    if (tid < 64) { float th = TWOPI_F * tid / 64.0f; c64[tid] = cosf(th); s64[tid] = sinf(th); }
    __syncthreads();
    int i = blockIdx.x * 256 + tid;                    // < 31744
    int mi = i % 31, j = (i / 31) & 63, b = i / 1984;
    int mu = mi - 15;
    const float* xp = x + (size_t)b * 4096 + (size_t)j * 64;
    float re = 0.f, im = 0.f;
    for (int t = 0; t < 64; ++t) {
        int u = (mu * t + 1024) & 63;
        re += xp[t] * c64[u];
        im -= xp[t] * s64[u];
    }
    size_t o = (size_t)(b * 64 + j) * 31 + mi;
    F[OFF_XMR + o] = re;
    F[OFF_XMI + o] = im;
}

// xh[b,l,mi] = sum_j D1F[l,j,mi] * xm[b,j,mi]
__global__ void k_xh(float* F) {
    int i = blockIdx.x * 256 + threadIdx.x;            // < 7936
    int mi = i % 31, l = (i / 31) & 15, b = i / 496;
    float xr = 0.f, xi = 0.f;
    for (int j = 0; j < 64; ++j) {
        float d = F[OFF_D1F + (size_t)(l * 64 + j) * 31 + mi];
        size_t o = (size_t)(b * 64 + j) * 31 + mi;
        xr += d * F[OFF_XMR + o];
        xi += d * F[OFF_XMI + o];
    }
    size_t o2 = (size_t)(b * 16 + l) * 31 + mi;
    F[OFF_XHR + o2] = xr;
    F[OFF_XHI + o2] = xi;
}

// Fused layer-1: per (b,f,j) plane:
//  P[m,n] = sum_l D1I[l,j,m,n] * xh[b,l,m] * conj(w1m[f,l,n])
//  r1[y,x] = relu(Re(IDFT2_32(P)));  s[y,k] = sum_x r1[y,x] e^{-2pi i (k-7) x/32}
__global__ void k_layer1(float* F, const float* __restrict__ w1r, const float* __restrict__ w1i) {
    int j = blockIdx.x, f = blockIdx.y, b = blockIdx.z;
    int tid = threadIdx.x;
    __shared__ float Ar[496], Ai[496], Br[496], Bi[496];
    __shared__ float Pr[961], Pi[961];
    __shared__ float Tr[992], Ti[992];
    __shared__ float r1p[1024];
    __shared__ float c32[32], s32[32];
    if (tid < 32) { float th = TWOPI_F * tid / 32.0f; c32[tid] = cosf(th); s32[tid] = sinf(th); }
    for (int i = tid; i < 496; i += 256) {
        int l = i / 31, mi = i % 31;
        size_t oa = (size_t)(b * 16 + l) * 31 + mi;
        Ar[i] = F[OFF_XHR + oa];
        Ai[i] = F[OFF_XHI + oa];
        int mm = mi - 15;
        bool ok = (mm >= -l && mm <= l);
        size_t ow = (size_t)(f * 16 + l) * 31 + mi;
        Br[i] = ok ? w1r[ow] : 0.f;
        Bi[i] = ok ? -w1i[ow] : 0.f;   // conj
    }
    __syncthreads();
    for (int i = tid; i < 961; i += 256) {
        int m = i / 31, n = i % 31;
        float pr = 0.f, pi = 0.f;
        for (int l = 0; l < 16; ++l) {
            float d  = F[OFF_D1I + ((size_t)(l * 32 + j) * 31 + m) * 31 + n];
            float ar = Ar[l * 31 + m], ai = Ai[l * 31 + m];
            float br = Br[l * 31 + n], bi = Bi[l * 31 + n];
            pr += d * (ar * br - ai * bi);
            pi += d * (ar * bi + ai * br);
        }
        Pr[i] = pr; Pi[i] = pi;
    }
    __syncthreads();
    for (int i = tid; i < 992; i += 256) {             // T[m,x] = sum_n P[m,n] e^{+i th}
        int m = i >> 5, xx = i & 31;
        float tr = 0.f, ti = 0.f;
        for (int n = 0; n < 31; ++n) {
            int u = ((n - 15) * xx + 512) & 31;
            float cr = c32[u], sn = s32[u];
            float prm = Pr[m * 31 + n], pim = Pi[m * 31 + n];
            tr += cr * prm - sn * pim;
            ti += cr * pim + sn * prm;
        }
        Tr[i] = tr; Ti[i] = ti;
    }
    __syncthreads();
    for (int i = tid; i < 1024; i += 256) {            // r1[y,x] = relu(Re sum_m e^{+i th} T[m,x])
        int y = i >> 5, xx = i & 31;
        float v = 0.f;
        for (int m = 0; m < 31; ++m) {
            int u = ((m - 15) * y + 512) & 31;
            v += c32[u] * Tr[m * 32 + xx] - s32[u] * Ti[m * 32 + xx];
        }
        r1p[i] = fmaxf(v, 0.f);
    }
    __syncthreads();
    size_t base = ((size_t)(b * 32 + f) * 32 + j) * 480;
    for (int i = tid; i < 480; i += 256) {             // s[y,k] (x-axis forward DFT)
        int y = i / 15, k = i % 15;
        float sr = 0.f, si = 0.f;
        for (int xx = 0; xx < 32; ++xx) {
            int u = ((k - 7) * xx + 512) & 31;
            float r = r1p[y * 32 + xx];
            sr += r * c32[u];
            si -= r * s32[u];
        }
        F[OFF_SR + base + i] = sr;
        F[OFF_SI + base + i] = si;
    }
}

// xmn[b,c,j,m,k] = sum_y e^{-2pi i (m-7) y/32} * s[b,c,j,y,k]
__global__ void k_xmn(float* F) {
    __shared__ float c32[32], s32[32];
    int tid = threadIdx.x;
    if (tid < 32) { float th = TWOPI_F * tid / 32.0f; c32[tid] = cosf(th); s32[tid] = sinf(th); }
    __syncthreads();
    int i = blockIdx.x * 256 + tid;                    // < 3686400
    int k = i % 15, m = (i / 15) % 15, j = (i / 225) & 31, c = (i / 7200) & 31, b = i / 230400;
    size_t sb = ((size_t)(b * 32 + c) * 32 + j) * 480;
    float re = 0.f, im = 0.f;
    for (int y = 0; y < 32; ++y) {
        int u = ((m - 7) * y + 512) & 31;
        float cr = c32[u], sn = s32[u];
        float sr = F[OFF_SR + sb + y * 15 + k];
        float si = F[OFF_SI + sb + y * 15 + k];
        re += cr * sr + sn * si;
        im += cr * si - sn * sr;
    }
    size_t o = ((size_t)(b * 32 + c) * 32 + j) * 225 + m * 15 + k;
    F[OFF_XMNR + o] = re;
    F[OFF_XMNI + o] = im;
}

// xh2[b,c,l,m,k] = sum_j D2F[l,j,m,k] * xmn[b,c,j,m,k]
__global__ void k_xh2(float* F) {
    int i = blockIdx.x * 256 + threadIdx.x;            // < 921600
    int k = i % 15, m = (i / 15) % 15, l = (i / 225) & 7, c = (i / 1800) & 31, b = i / 57600;
    float xr = 0.f, xi = 0.f;
    for (int j = 0; j < 32; ++j) {
        float d = F[OFF_D2F + ((size_t)(l * 32 + j) * 15 + m) * 15 + k];
        size_t o = ((size_t)(b * 32 + c) * 32 + j) * 225 + m * 15 + k;
        xr += d * F[OFF_XMNR + o];
        xi += d * F[OFF_XMNI + o];
    }
    size_t o2 = ((size_t)(b * 32 + c) * 8 + l) * 225 + m * 15 + k;
    F[OFF_X2R + o2] = xr;
    F[OFF_X2I + o2] = xi;
}

// B-fragment pack: W[K,f] = sum_n D2I0[l,m,n] * w2{r,i}[c,f,l,n,k] * mask2(l,n,k), as f16
// K = ((c*8+l)*15+m)*15+k)*2 + reim ; CDNA5 B layout: lane = (f&15)+16*((K>>3)&1), half = (K&7)|((K>>4)<<3)
__global__ void k_packW(const float* F, const float* __restrict__ w2r,
                        const float* __restrict__ w2i, _Float16* __restrict__ WP) {
    int i = blockIdx.x * 256 + threadIdx.x;            // < 7372800
    int h = i & 15, L = (i >> 4) & 31, tt = (i >> 9) % 3600, w = (i >> 9) / 3600;
    int K = 32 * tt + ((h & 7) | ((L >> 4) << 3) | ((h >> 3) << 4));
    int f = 16 * w + (L & 15);
    int reim = K & 1, q = K >> 1;
    int k = q % 15, m = (q / 15) % 15, l = (q / 225) & 7, c = q / 1800;
    float val = 0.f;
    if (k - 7 >= -l && k - 7 <= l) {
        const float* wsrc = reim ? w2i : w2r;
        size_t wb = ((size_t)(c * 64 + f) * 8 + l) * 225;
        for (int n = 0; n < 15; ++n)
            val += F[OFF_D2I0 + (size_t)(l * 15 + m) * 15 + n] * wsrc[wb + n * 15 + k];
    }
    WP[i] = (_Float16)val;
}

// A-fragment pack: X[b,K] = Re/Im(xh2[b,c,l,m,k]) as f16
// CDNA5 A layout: lane = b + 16*((K>>3)&1), half = (K&7)|((K>>4)<<3)
__global__ void k_packX(const float* F, _Float16* __restrict__ XP) {
    int i = blockIdx.x * 256 + threadIdx.x;            // < 1843200
    int h = i & 15, L = (i >> 4) & 31, tt = i >> 9;
    int Kloc = (h & 7) | ((L >> 4) << 3) | ((h >> 3) << 4);
    int K = 32 * tt + Kloc;
    int b = L & 15;
    int reim = K & 1, q = K >> 1;
    int k = q % 15, m = (q / 15) % 15, l = (q / 225) & 7, c = q / 1800;
    size_t o = ((size_t)(b * 32 + c) * 8 + l) * 225 + m * 15 + k;
    XP[i] = (_Float16)(reim ? F[OFF_X2I + o] : F[OFF_X2R + o]);
}

// feat = relu(X(16x115200) * W(115200x64)), one wave per 16x16 N-tile, WMMA f16->f32
__global__ void k_gemm(const _Float16* __restrict__ XP, const _Float16* __restrict__ WP,
                       float* __restrict__ feat) {
    int wave = threadIdx.x >> 5;
    int lane = threadIdx.x & 31;
    v8f acc = {0.f, 0.f, 0.f, 0.f, 0.f, 0.f, 0.f, 0.f};
    const _Float16* xb = XP + lane * 16;
    const _Float16* wb = WP + (size_t)wave * 3600 * 512 + lane * 16;
    for (int t = 0; t < 3600; ++t) {
        __builtin_prefetch(xb + (t + 8) * 512, 0, 0);
        __builtin_prefetch(wb + (t + 8) * 512, 0, 0);
        v16h a = *(const v16h*)(xb + (size_t)t * 512);
        v16h bfrag = *(const v16h*)(wb + (size_t)t * 512);
        acc = __builtin_amdgcn_wmma_f32_16x16x32_f16(false, a, false, bfrag,
                                                     (short)0, acc, false, false);
    }
    int M0 = (lane >> 4) * 8;
    int N = wave * 16 + (lane & 15);
#pragma unroll
    for (int r = 0; r < 8; ++r)
        feat[(M0 + r) * 64 + N] = fmaxf(acc[r], 0.0f);
}

// head: conv1d(8) + bias + relu -> BN(batch stats) -> FC
__global__ void k_head(const float* F, const float* __restrict__ c3w, const float* __restrict__ c3b,
                       const float* __restrict__ bng, const float* __restrict__ bnb,
                       const float* __restrict__ fcw, const float* __restrict__ fcb,
                       float* __restrict__ out) {
    __shared__ float sFeat[16 * 64];
    __shared__ float sR3[16 * 570];
    __shared__ float sScale[10], sShift[10];
    int tid = threadIdx.x;
    for (int i = tid; i < 1024; i += 256) sFeat[i] = F[OFF_FEAT + i];
    __syncthreads();
    for (int i = tid; i < 16 * 570; i += 256) {
        int b = i / 570, r = i % 570, ch = r / 57, p = r % 57;
        float a = c3b[ch];
        for (int q = 0; q < 8; ++q) a += sFeat[b * 64 + p + q] * c3w[ch * 8 + q];
        sR3[i] = fmaxf(a, 0.f);
    }
    __syncthreads();
    if (tid < 10) {
        float s = 0.f, s2 = 0.f;
        for (int b = 0; b < 16; ++b)
            for (int p = 0; p < 57; ++p) {
                float v = sR3[b * 570 + tid * 57 + p];
                s += v; s2 += v * v;
            }
        float mu = s / 912.f;
        float var = s2 / 912.f - mu * mu;
        float sc = bng[tid] * rsqrtf(var + 1e-5f);
        sScale[tid] = sc;
        sShift[tid] = bnb[tid] - mu * sc;
    }
    __syncthreads();
    for (int i = tid; i < 160; i += 256) {
        int b = i / 10, o = i % 10;
        float a = fcb[o];
        for (int ch = 0; ch < 10; ++ch) {
            float sc = sScale[ch], sh = sShift[ch];
            const float* fw = fcw + (size_t)o * 570 + ch * 57;
            const float* rr = sR3 + b * 570 + ch * 57;
            for (int p = 0; p < 57; ++p) a += (rr[p] * sc + sh) * fw[p];
        }
        out[i] = a;
    }
}

// ---------------- launcher ----------------
extern "C" void kernel_launch(void* const* d_in, const int* in_sizes, int n_in,
                              void* d_out, int out_size, void* d_ws, size_t ws_size,
                              hipStream_t stream) {
    (void)in_sizes; (void)n_in; (void)out_size; (void)ws_size;
    const float* x   = (const float*)d_in[0];
    const float* w1r = (const float*)d_in[1];
    const float* w1i = (const float*)d_in[2];
    const float* w2r = (const float*)d_in[3];
    const float* w2i = (const float*)d_in[4];
    const float* c3w = (const float*)d_in[5];
    const float* c3b = (const float*)d_in[6];
    const float* bng = (const float*)d_in[7];
    const float* bnb = (const float*)d_in[8];
    const float* fcw = (const float*)d_in[9];
    const float* fcb = (const float*)d_in[10];
    float* outp = (float*)d_out;

    double*   Dd = (double*)d_ws;
    float*    F  = (float*)d_ws + 640;
    _Float16* XP = (_Float16*)(F + OFF_XP);
    _Float16* WP = (_Float16*)(F + OFF_WP);

    // constants (deterministic rebuild each call; graph-capture safe)
    k_scalars<<<1, 64, 0, stream>>>(Dd);
    k_zero<<<2279, 256, 0, stream>>>(F, 583176);       // zero D-tables (masked regions)
    k_d1f <<<124,  256, 0, stream>>>(Dd, F);
    k_d1i <<<1922, 256, 0, stream>>>(Dd, F);
    k_d2f <<<225,  256, 0, stream>>>(Dd, F);
    k_d2i0<<<8,    256, 0, stream>>>(Dd, F);

    // forward
    k_fft1  <<<124, 256, 0, stream>>>(x, F);
    k_xh    <<<31,  256, 0, stream>>>(F);
    k_layer1<<<dim3(32, 32, 16), 256, 0, stream>>>(F, w1r, w1i);
    k_xmn   <<<14400, 256, 0, stream>>>(F);
    k_xh2   <<<3600,  256, 0, stream>>>(F);
    k_packW <<<28800, 256, 0, stream>>>(F, w2r, w2i, WP);
    k_packX <<<7200,  256, 0, stream>>>(F, XP);
    k_gemm  <<<1, 128, 0, stream>>>(XP, WP, F + OFF_FEAT);
    k_head  <<<1, 256, 0, stream>>>(F, c3w, c3b, bng, bnb, fcw, fcb, outp);
}
